// AutoEncoderTopK_39676907880699
// MI455X (gfx1250) — matrix operand order repaired and checked by reading, not compiled, not run on hardware
//
#include <hip/hip_runtime.h>
#include <hip/hip_bf16.h>

// ---------------------------------------------------------------------------
// AutoEncoderTopK (sparse autoencoder forward) for MI455X / gfx1250.
//   encode : pre = (x - b_dec) @ W_enc^T + b_enc ; acts = relu(pre)
//   top-64 per row ; decode sparsely: x_hat = b_dec + sum_j v_j * W_dec[:,i_j]
// Encoder GEMM: v_wmma_f32_16x16x32_bf16, 32x64 tile per wave, ping-pong
// register double buffering (2x unrolled K loop, no inter-buffer copies).
// ---------------------------------------------------------------------------

#define ACT_DIM   768
#define DICT_SIZE 24576
#define BATCH     4096
#define TOPK      64

typedef __bf16 bf16x8  __attribute__((ext_vector_type(8)));
typedef __bf16 bf16x16 __attribute__((ext_vector_type(16)));
typedef float  floatx8 __attribute__((ext_vector_type(8)));

// ---------------------------------------------------------------- converts --
__global__ void __launch_bounds__(256) k_prep_x(const float* __restrict__ x,
                                                const float* __restrict__ b_dec,
                                                __bf16* __restrict__ xb, int n) {
    int i = blockIdx.x * 256 + threadIdx.x;
    if (i < n) {
        int d = i % ACT_DIM;
        xb[i] = (__bf16)(x[i] - b_dec[d]);
    }
}

__global__ void __launch_bounds__(256) k_prep_w(const float* __restrict__ w,
                                                __bf16* __restrict__ wb, int n) {
    int i = blockIdx.x * 256 + threadIdx.x;
    if (i < n) wb[i] = (__bf16)w[i];
}

// W_dec [768][24576] row-major -> wdt [24576][768] row-major (coalesced both ways)
__global__ void __launch_bounds__(256) k_transpose(const float* __restrict__ wdec,
                                                   float* __restrict__ wdt) {
    __shared__ float tile[32][33];
    int f0 = blockIdx.x * 32, d0 = blockIdx.y * 32;
    int tx = threadIdx.x & 31, ty = threadIdx.x >> 5;   // 32 x 8
    #pragma unroll
    for (int r = ty; r < 32; r += 8)
        tile[r][tx] = wdec[(size_t)(d0 + r) * DICT_SIZE + f0 + tx];
    __syncthreads();
    #pragma unroll
    for (int r = ty; r < 32; r += 8)
        wdt[(size_t)(f0 + r) * ACT_DIM + d0 + tx] = tile[tx][r];
}

// ------------------------------------------------------- encoder WMMA GEMM --
// acts[B][F] = relu( xb[B][K] * wb[F][K]^T + b_enc[F] ), K = 768.
// Block = 256 threads = 8 waves arranged 4(M) x 2(N); block tile 128x128.
// Wave tile 32(M) x 64(N): 8 f32 accumulators of 16x16, 8 WMMAs per K-step.
static __device__ __forceinline__ bf16x16 load_afrag(const __bf16* p) {
    // 16-bit A layout (contiguous, matching B): lanes 0-15 hold K k..k+15,
    // lanes 16-31 hold K k+16..k+31 -> two contiguous 16B loads.
    bf16x8 lo = *(const bf16x8*)p;
    bf16x8 hi = *(const bf16x8*)(p + 8);
    return __builtin_shufflevector(lo, hi,
               0,1,2,3,4,5,6,7,8,9,10,11,12,13,14,15);
}

__global__ void __launch_bounds__(256) k_enc_gemm(const __bf16* __restrict__ xb,
                                                  const __bf16* __restrict__ wb,
                                                  const float*  __restrict__ b_enc,
                                                  float* __restrict__ acts) {
    const int lane = threadIdx.x & 31;
    const int wave = threadIdx.x >> 5;
    const int wm   = wave & 3;                         // 4 waves along M
    const int wn   = wave >> 2;                        // 2 waves along N
    const int m0   = blockIdx.y * 128 + wm * 32;       // 4096/128  = 32 in y
    const int f0   = blockIdx.x * 128 + wn * 64;       // 24576/128 = 192 in x
    const int r    = lane & 15;
    const int grp  = lane >> 4;

    floatx8 acc[2][4] = {};

    const __bf16* arow0 = xb + (size_t)(m0 + r) * ACT_DIM + grp * 16;
    const __bf16* arow1 = arow0 + (size_t)16 * ACT_DIM;
    // B (32x16 K x N) tile of wb^T: lane holds col f0+n*16+r, 16 contiguous K.
    const __bf16* bbase = wb + (size_t)(f0 + r) * ACT_DIM + grp * 16;

    // Ping-pong fragment buffers: no register copies, names alternate.
    bf16x16 a0[2], b0[4], a1[2], b1[4];

    a0[0] = load_afrag(arow0);
    a0[1] = load_afrag(arow1);
    #pragma unroll
    for (int n = 0; n < 4; ++n)
        b0[n] = *(const bf16x16*)(bbase + (size_t)n * 16 * ACT_DIM);

    #pragma unroll 1
    for (int kk = 0; kk < ACT_DIM; kk += 64) {          // 12 iterations
        const int k1 = kk + 32;
        // load buffer-1 (k1), then compute on buffer-0 (kk)
        a1[0] = load_afrag(arow0 + k1);
        a1[1] = load_afrag(arow1 + k1);
        #pragma unroll
        for (int n = 0; n < 4; ++n)
            b1[n] = *(const bf16x16*)(bbase + (size_t)n * 16 * ACT_DIM + k1);

        #pragma unroll
        for (int mi = 0; mi < 2; ++mi)
            #pragma unroll
            for (int n = 0; n < 4; ++n)
                acc[mi][n] = __builtin_amdgcn_wmma_f32_16x16x32_bf16(
                                 false, a0[mi], false, b0[n], (short)0,
                                 acc[mi][n], false, false);

        const int k0 = (kk + 64 < ACT_DIM) ? kk + 64 : 0;   // wrap: harmless
        // load buffer-0 (k0), then compute on buffer-1 (k1)
        a0[0] = load_afrag(arow0 + k0);
        a0[1] = load_afrag(arow1 + k0);
        #pragma unroll
        for (int n = 0; n < 4; ++n)
            b0[n] = *(const bf16x16*)(bbase + (size_t)n * 16 * ACT_DIM + k0);

        #pragma unroll
        for (int mi = 0; mi < 2; ++mi)
            #pragma unroll
            for (int n = 0; n < 4; ++n)
                acc[mi][n] = __builtin_amdgcn_wmma_f32_16x16x32_bf16(
                                 false, a1[mi], false, b1[n], (short)0,
                                 acc[mi][n], false, false);
    }

    // C/D layout: VGPR v, lanes 0-15 -> M=v, N=lane ; lanes 16-31 -> M=8+v.
    #pragma unroll
    for (int n = 0; n < 4; ++n) {
        int col  = f0 + n * 16 + r;
        float be = b_enc[col];
        #pragma unroll
        for (int mi = 0; mi < 2; ++mi) {
            #pragma unroll
            for (int v = 0; v < 8; ++v) {
                int m = m0 + mi * 16 + grp * 8 + v;
                float val = acc[mi][n][v] + be;
                acts[(size_t)m * DICT_SIZE + col] = val > 0.f ? val : 0.f;
            }
        }
    }
}

// ------------------------------------------------------------------ top-64 --
static __device__ __forceinline__ unsigned long long
shfl_xor_u64(unsigned long long x, int m) {
    unsigned int lo = (unsigned int)x, hi = (unsigned int)(x >> 32);
    lo = __shfl_xor(lo, m, 32);
    hi = __shfl_xor(hi, m, 32);
    return ((unsigned long long)hi << 32) | lo;
}

// One 1024-thread block per row. Each thread holds 24 acts in registers.
// Key = ((bits(v)+1) << 32) | ~idx : descending value, ascending index on ties.
__global__ void __launch_bounds__(1024) k_topk(const float* __restrict__ acts,
                                               float* __restrict__ tvals,
                                               int*   __restrict__ tidx) {
    const int row  = blockIdx.x;
    const int tid  = threadIdx.x;
    const int lane = tid & 31, wv = tid >> 5;
    const float* arow = acts + (size_t)row * DICT_SIZE;

    float v[24];
    #pragma unroll
    for (int j = 0; j < 24; ++j) v[j] = arow[j * 1024 + tid];   // coalesced

    __shared__ unsigned long long warp_red[32];
    __shared__ unsigned long long winner;

    for (int it = 0; it < TOPK; ++it) {
        unsigned long long best = 0ull;
        #pragma unroll
        for (int j = 0; j < 24; ++j) {
            unsigned int bits = __float_as_uint(v[j]);      // relu => >= +0
            unsigned long long cand = (bits & 0x80000000u) ? 0ull
                : ((((unsigned long long)(bits + 1u)) << 32)
                   | (unsigned int)~(unsigned int)(j * 1024 + tid));
            best = cand > best ? cand : best;
        }
        #pragma unroll
        for (int s = 16; s; s >>= 1) {
            unsigned long long o = shfl_xor_u64(best, s);
            best = o > best ? o : best;
        }
        if (lane == 0) warp_red[wv] = best;
        __syncthreads();
        if (wv == 0) {
            unsigned long long b2 = warp_red[lane];
            #pragma unroll
            for (int s = 16; s; s >>= 1) {
                unsigned long long o = shfl_xor_u64(b2, s);
                b2 = o > b2 ? o : b2;
            }
            if (lane == 0) {
                winner = b2;
                unsigned int key = (unsigned int)(b2 >> 32);
                unsigned int idx = ~(unsigned int)b2;
                tvals[row * TOPK + it] = key ? __uint_as_float(key - 1u) : 0.f;
                tidx [row * TOPK + it] = key ? (int)idx : 0;
            }
        }
        __syncthreads();
        unsigned long long w = winner;
        if ((unsigned int)(w >> 32)) {
            unsigned int widx = ~(unsigned int)w;
            #pragma unroll
            for (int j = 0; j < 24; ++j)
                if (widx == (unsigned int)(j * 1024 + tid)) v[j] = -1.0f;
        }
        __syncthreads();
    }
}

// ----------------------------------------------------------- sparse decode --
// x_hat[row][d] = b_dec[d] + sum_j v_j * wdt[idx_j][d] ; block per row.
// Data-dependent gather: prefetch next row (global_prefetch_b8) while FMA-ing.
__global__ void __launch_bounds__(256) k_decode(const float* __restrict__ tvals,
                                                const int*   __restrict__ tidx,
                                                const float* __restrict__ wdt,
                                                const float* __restrict__ b_dec,
                                                float* __restrict__ out) {
    const int row = blockIdx.x;
    const int tid = threadIdx.x;
    __shared__ float sv[TOPK];
    __shared__ int   si[TOPK];
    if (tid < TOPK) {
        sv[tid] = tvals[row * TOPK + tid];
        si[tid] = tidx [row * TOPK + tid];
    }
    __syncthreads();
    float a0 = 0.f, a1 = 0.f, a2 = 0.f;
    for (int j = 0; j < TOPK; ++j) {
        if (j + 1 < TOPK)
            __builtin_prefetch(wdt + (size_t)si[j + 1] * ACT_DIM + tid, 0, 1);
        float vj = sv[j];
        if (vj > 0.f) {
            const float* wrow = wdt + (size_t)si[j] * ACT_DIM;
            a0 += vj * wrow[tid];
            a1 += vj * wrow[tid + 256];
            a2 += vj * wrow[tid + 512];
        }
    }
    out[(size_t)row * ACT_DIM + tid      ] = a0 + b_dec[tid      ];
    out[(size_t)row * ACT_DIM + tid + 256] = a1 + b_dec[tid + 256];
    out[(size_t)row * ACT_DIM + tid + 512] = a2 + b_dec[tid + 512];
}

// ----------------------------------------------------------------- launch ---
extern "C" void kernel_launch(void* const* d_in, const int* in_sizes, int n_in,
                              void* d_out, int out_size, void* d_ws, size_t ws_size,
                              hipStream_t stream) {
    const float* x     = (const float*)d_in[0];   // [4096, 768]
    const float* W_enc = (const float*)d_in[1];   // [24576, 768]
    const float* b_enc = (const float*)d_in[2];   // [24576]
    const float* W_dec = (const float*)d_in[3];   // [768, 24576]
    const float* b_dec = (const float*)d_in[4];   // [768]
    float* out = (float*)d_out;                   // [4096, 768]

    char* ws = (char*)d_ws;
    __bf16* xb    = (__bf16*)(ws);                         //  6,291,456 B
    __bf16* wb    = (__bf16*)(ws + 6291456ull);            // 37,748,736 B
    float*  wdt   = (float*) (ws + 44040192ull);           // 75,497,472 B
    float*  tvals = (float*) (ws + 119537664ull);          //  1,048,576 B
    int*    tidx  = (int*)   (ws + 120586240ull);          //  1,048,576 B
    float*  acts  = (float*) (ws + 121634816ull);          // 402,653,184 B

    const int nx = BATCH * ACT_DIM;
    const int nw = DICT_SIZE * ACT_DIM;
    k_prep_x<<<(nx + 255) / 256, 256, 0, stream>>>(x, b_dec, xb, nx);
    k_prep_w<<<(nw + 255) / 256, 256, 0, stream>>>(W_enc, wb, nw);
    k_transpose<<<dim3(DICT_SIZE / 32, ACT_DIM / 32), 256, 0, stream>>>(W_dec, wdt);
    k_enc_gemm<<<dim3(DICT_SIZE / 128, BATCH / 128), 256, 0, stream>>>(xb, wb, b_enc, acts);
    k_topk<<<BATCH, 1024, 0, stream>>>(acts, tvals, tidx);
    k_decode<<<BATCH, 256, 0, stream>>>(tvals, tidx, wdt, b_dec, out);
}
